// MyModel_Attn_75436805587655
// MI455X (gfx1250) — matrix operand order repaired and checked by reading, not compiled, not run on hardware
//
#include <hip/hip_runtime.h>
#include <cstdint>
#include <cstddef>

// ---------------- model dims ----------------
#define B_    16
#define T_    1024
#define E_    256
#define S_    1040   // T + N_OUT
#define NH_   4
#define DK_   64
#define G_    1024   // 4*E gates
#define NOUT_ 16
#define SP_   1056   // S padded to multiple of 32 for WMMA K-loop

typedef _Float16 f16;
typedef __attribute__((ext_vector_type(16))) _Float16 v16h;
typedef __attribute__((ext_vector_type(8)))  _Float16 v8h;
typedef __attribute__((ext_vector_type(8)))  float    v8f;

#define WMMA_F16(a, b, c) \
  __builtin_amdgcn_wmma_f32_16x16x32_f16(false, (a), false, (b), (short)0, (c), false, false)

// ---- branch-free fast transcendentals (keep the sequential LSTM path short) ----
#define LOG2E_ 1.4426950408889634f
__device__ __forceinline__ float fast_exp(float x) {
#if __has_builtin(__builtin_amdgcn_exp2f)
  return __builtin_amdgcn_exp2f(x * LOG2E_);
#else
  return __expf(x);
#endif
}
__device__ __forceinline__ float fast_rcp(float x) {
#if __has_builtin(__builtin_amdgcn_rcpf)
  return __builtin_amdgcn_rcpf(x);
#else
  return 1.f / x;
#endif
}
// sigmoid(x) = 1/(1+e^-x); x->-inf: exp->+inf -> rcp->0 ; x->+inf: ->1   (no branches)
__device__ __forceinline__ float fast_sigmoid(float x) { return fast_rcp(1.f + fast_exp(-x)); }
// tanh(x) = 1 - 2/(1+e^{2x}); x->+inf: ->1 ; x->-inf: ->-1               (no branches)
__device__ __forceinline__ float fast_tanh(float x) {
#if __has_builtin(__builtin_amdgcn_tanhf)
  return __builtin_amdgcn_tanhf(x);
#else
  return 1.f - 2.f * fast_rcp(1.f + fast_exp(2.f * x));
#endif
}

__device__ __forceinline__ v16h make_frag(v8h lo, v8h hi) {
  union { f16 s[16]; v16h v; } u;
#pragma unroll
  for (int i = 0; i < 8; ++i) { u.s[i] = lo[i]; u.s[8 + i] = hi[i]; }
  return u.v;
}

// A fragment (16xK=32) from row-major [rows x lda] f16 matrix.
// Lanes 0-15: row M=lane, K = k0+0..7 (V0-3) and k0+16..23 (V4-7); lanes 16-31: K = +8.
__device__ __forceinline__ v16h load_frag_a(const f16* A, int lda, int row0, int k0, int lane) {
  const f16* base = A + (size_t)(row0 + (lane & 15)) * lda + k0 + ((lane >> 4) << 3);
  v8h lo = *(const v8h*)(base);
  v8h hi = *(const v8h*)(base + 16);
  return make_frag(lo, hi);
}
// B fragment (K=32 x N=16) from row-major [N x lda] matrix (i.e. B[n][k], C = A*B^T style).
// Lanes 0-15: col N=lane, K = k0+0..7 and k0+16..23; lanes 16-31: K = +8.
__device__ __forceinline__ v16h load_frag_b(const f16* Bm, int lda, int n0, int k0, int lane) {
  const f16* base = Bm + (size_t)(n0 + (lane & 15)) * lda + k0 + ((lane >> 4) << 3);
  v8h lo = *(const v8h*)(base);
  v8h hi = *(const v8h*)(base + 16);
  return make_frag(lo, hi);
}

// ---------------- precompute (tab path + algebraic reductions), 1 block ----------------
__global__ __launch_bounds__(256) void precompute_kernel(
    const float* __restrict__ x_tab, const float* __restrict__ tabW, const float* __restrict__ tabb,
    const float* __restrict__ tabW1, const float* __restrict__ tabb1,
    const float* __restrict__ tabW2, const float* __restrict__ tabb2,
    const float* __restrict__ tabW3, const float* __restrict__ tabb3,
    const float* __restrict__ tsW, const float* __restrict__ tsb,
    const float* __restrict__ eWih0, const float* __restrict__ ebih0, const float* __restrict__ ebhh0,
    const float* __restrict__ fW, const float* __restrict__ fb,
    const float* __restrict__ dWih0, const float* __restrict__ dbih0, const float* __restrict__ dbhh0,
    const float* __restrict__ ebih1, const float* __restrict__ ebhh1,
    const float* __restrict__ dbih1, const float* __restrict__ dbhh1,
    float* __restrict__ t2, float* __restrict__ mlp1, float* __restrict__ mlp2,
    float* __restrict__ st, float* __restrict__ A0, float* __restrict__ c0,
    float* __restrict__ afut, float* __restrict__ cfut,
    float* __restrict__ b1e, float* __restrict__ b1d) {
  const int tid = threadIdx.x;
  // t2[b, f*256+e] = (x_tab[b,0]*tabW[0,f]+tabb[0,f]) * tabW[f,e] + tabb[f,e]
  for (int i = tid; i < 16 * 4096; i += 256) {
    int b = i >> 12, c = i & 4095, f = c >> 8;
    float t1f = x_tab[b * 16 + 0] * tabW[f] + tabb[f];
    t2[i] = t1f * tabW[c] + tabb[c];
  }
  __syncthreads();
  for (int i = tid; i < 16 * 64; i += 256) {
    int b = i >> 6, o = i & 63;
    float s = tabb1[o];
    const float* row = t2 + (size_t)b * 4096;
    const float* w = tabW1 + (size_t)o * 4096;
    for (int j = 0; j < 4096; ++j) s += row[j] * w[j];
    mlp1[i] = fmaxf(s, 0.f);
  }
  __syncthreads();
  for (int i = tid; i < 16 * 32; i += 256) {
    int b = i >> 5, o = i & 31;
    float s = tabb2[o];
    for (int j = 0; j < 64; ++j) s += mlp1[b * 64 + j] * tabW2[o * 64 + j];
    mlp2[i] = fmaxf(s, 0.f);
  }
  __syncthreads();
  for (int i = tid; i < 16 * 512; i += 256) {  // st flat == torch .view(L,B,E)
    int b = i >> 9, o = i & 511;
    float s = tabb3[o];
    for (int j = 0; j < 32; ++j) s += mlp2[b * 32 + j] * tabW3[o * 32 + j];
    st[i] = s;
  }
  // reduced input projection for encoder L0: A0[g,f], c0[g]
  for (int i = tid; i < G_ * 8; i += 256) {
    int g = i >> 3, f = i & 7;
    float s = 0.f;
    const float* w = eWih0 + (size_t)g * 2048 + f * 256;
    const float* tw = tsW + f * 256;
    for (int e = 0; e < 256; ++e) s += tw[e] * w[e];
    A0[i] = s;
  }
  for (int g = tid; g < G_; g += 256) {
    float s = ebih0[g] + ebhh0[g];
    const float* w = eWih0 + (size_t)g * 2048;
    for (int j = 0; j < 2048; ++j) s += tsb[j] * w[j];
    c0[g] = s;
    float sa = 0.f, sc = dbih0[g] + dbhh0[g];
    const float* wd = dWih0 + (size_t)g * 256;
    for (int e = 0; e < 256; ++e) { sa += fW[e] * wd[e]; sc += fb[e] * wd[e]; }
    afut[g] = sa; cfut[g] = sc;
    b1e[g] = ebih1[g] + ebhh1[g];
    b1d[g] = dbih1[g] + dbhh1[g];
  }
}

// ---------------- weight conversion / packing ----------------
__global__ void f32_to_f16_kernel(const float* __restrict__ src, f16* __restrict__ dst, int n) {
  int i = blockIdx.x * 256 + threadIdx.x;
  if (i < n) dst[i] = (f16)src[i];
}
__global__ void pack_w_kernel(const float* __restrict__ Wih, const float* __restrict__ Whh,
                              f16* __restrict__ Wp, int Kin, int n) {
  int i = blockIdx.x * 256 + threadIdx.x;
  if (i < n) {
    int Kt = Kin + 256;
    int g = i / Kt, c = i % Kt;
    float v = (c < Kin) ? Wih[(size_t)g * Kin + c] : Whh[(size_t)g * 256 + (c - Kin)];
    Wp[i] = (f16)v;
  }
}

// ---------------- sequential LSTM layer: one workgroup, 16 waves, WMMA per step ----------
// gates(16x1024) = [X_t | h] @ Wpack^T + bias (+ reduced input Σ_f x[b,t,f]*Ared[g,f]).
// Wave w owns e-range [16w,16w+16) and its 4 gate tiles (i,f,g,o), so activations and the
// cell state c stay entirely in registers; h is shared via a double-buffered LDS panel
// (one s_barrier per step: step t reads panel t&1, writes h/X_{t+1} into panel (t+1)&1).
__global__ __launch_bounds__(512) void lstm_seq_kernel(
    const f16* __restrict__ Xbf, int Kin, int x_rows,
    const float* __restrict__ xraw, int F, int Tlen,
    const float* __restrict__ Ared, const float* __restrict__ bias,
    const f16* __restrict__ W,                       // (1024, Kin+256) f16
    const float* __restrict__ h_init, const float* __restrict__ c_init,  // (16,256)
    f16* __restrict__ ybf, int ybf_rows,
    float* __restrict__ yf32, int yf_rows, int yf_toff,
    float* __restrict__ h_out, float* __restrict__ c_out) {
  __shared__ f16 Abuf[2][16][512];
  __shared__ float xs[2][16][8];
  const int tid = threadIdx.x;
  const int lane = tid & 31;
  const int wave = tid >> 5;            // 0..15
  const int ebase = wave << 4;
  const int ncol = ebase + (lane & 15); // e
  const int mrow = (lane >> 4) << 3;    // batch row base (0 or 8)
  const int Ktot = Kin + E_;

  float biasv[4];
  float AredL[4][8];
#pragma unroll
  for (int gt = 0; gt < 4; ++gt) {
    int g = gt * E_ + ncol;
    biasv[gt] = bias[g];
    for (int f = 0; f < F; ++f) AredL[gt][f] = Ared[(size_t)g * F + f];
  }
  float creg[8];
#pragma unroll
  for (int r = 0; r < 8; ++r) creg[r] = c_init[(size_t)(r + mrow) * E_ + ncol];

  for (int i = tid; i < 16 * E_; i += 512)
    Abuf[0][i >> 8][Kin + (i & 255)] = (f16)h_init[i];
  if (Xbf)
    for (int i = tid; i < 16 * Kin; i += 512) {
      int b = i / Kin, c = i % Kin;
      Abuf[0][b][c] = Xbf[((size_t)b * x_rows + 0) * Kin + c];
    }
  if (F > 0 && tid < 16 * F) {
    int b = tid / F, f = tid % F;
    xs[0][b][f] = xraw[((size_t)b * Tlen + 0) * F + f];
  }
  __syncthreads();

  float hlast[8];
  for (int t = 0; t < Tlen; ++t) {
    const int p = t & 1;
    v8f acc[4];
#pragma unroll
    for (int gt = 0; gt < 4; ++gt) {
      float v = biasv[gt];
#pragma unroll
      for (int r = 0; r < 8; ++r) acc[gt][r] = v;
    }
    if (F > 0) {
#pragma unroll
      for (int r = 0; r < 8; ++r) {
        int b = r + mrow;
        for (int f = 0; f < F; ++f) {
          float xv = xs[p][b][f];
#pragma unroll
          for (int gt = 0; gt < 4; ++gt) acc[gt][r] += xv * AredL[gt][f];
        }
      }
    }
    for (int kk = 0; kk < Ktot; kk += 32) {
      v16h a = load_frag_a(&Abuf[p][0][0], 512, 0, kk, lane);
#pragma unroll
      for (int gt = 0; gt < 4; ++gt) {
        v16h b = load_frag_b(W, Ktot, gt * E_ + ebase, kk, lane);
        acc[gt] = WMMA_F16(a, b, acc[gt]);
      }
    }
#pragma unroll
    for (int r = 0; r < 8; ++r) {
      float iv = fast_sigmoid(acc[0][r]);
      float fv = fast_sigmoid(acc[1][r]);
      float gv = fast_tanh(acc[2][r]);
      float ov = fast_sigmoid(acc[3][r]);
      float cc = fv * creg[r] + iv * gv;
      creg[r] = cc;
      hlast[r] = ov * fast_tanh(cc);
    }
    // write new h (and next X) into the *other* panel; nobody reads it until the barrier
    const int q = p ^ 1;
#pragma unroll
    for (int r = 0; r < 8; ++r) {
      f16 hb = (f16)hlast[r];
      int b = r + mrow;
      Abuf[q][b][Kin + ncol] = hb;
      if (ybf)  ybf[((size_t)b * ybf_rows + t) * E_ + ncol] = hb;
      if (yf32) yf32[((size_t)b * yf_rows + (t + yf_toff)) * E_ + ncol] = hlast[r];
    }
    if (t + 1 < Tlen) {
      if (Xbf)
        for (int i = tid; i < 16 * Kin; i += 512) {
          int b = i / Kin, c = i % Kin;
          Abuf[q][b][c] = Xbf[((size_t)b * x_rows + (t + 1)) * Kin + c];
        }
      if (F > 0 && tid < 16 * F) {
        int b = tid / F, f = tid % F;
        xs[q][b][f] = xraw[((size_t)b * Tlen + (t + 1)) * F + f];
      }
    }
    __syncthreads();
  }
#pragma unroll
  for (int r = 0; r < 8; ++r) {
    h_out[(size_t)(r + mrow) * E_ + ncol] = hlast[r];
    c_out[(size_t)(r + mrow) * E_ + ncol] = creg[r];
  }
}

// ---------------- layernorm: one block per row, write f16 ----------------
__global__ __launch_bounds__(256) void layernorm_kernel(
    const float* __restrict__ x, const float* __restrict__ g, const float* __restrict__ b,
    f16* __restrict__ out) {
  __shared__ float red[256];
  const int row = blockIdx.x, tid = threadIdx.x;
  float v = x[(size_t)row * E_ + tid];
  red[tid] = v; __syncthreads();
  for (int o = 128; o > 0; o >>= 1) { if (tid < o) red[tid] += red[tid + o]; __syncthreads(); }
  float m = red[0] * (1.f / E_); __syncthreads();
  float d = v - m;
  red[tid] = d * d; __syncthreads();
  for (int o = 128; o > 0; o >>= 1) { if (tid < o) red[tid] += red[tid + o]; __syncthreads(); }
  float var = red[0] * (1.f / E_);
  out[(size_t)row * E_ + tid] = (f16)(d * rsqrtf(var + 1e-5f) * g[tid] + b[tid]);
}

// ---------------- generic C = A*B^T (+bias) f16 WMMA GEMM ----------------
__global__ __launch_bounds__(128) void gemm_f16_kernel(
    const f16* __restrict__ A, const f16* __restrict__ Bm, const float* __restrict__ bias,
    f16* __restrict__ Cbf, float* __restrict__ Cf, int N, int K) {
  const int lane = threadIdx.x & 31;
  const int wave = threadIdx.x >> 5;
  const int m0 = blockIdx.x << 4;
  const int n0 = (blockIdx.y << 6) + (wave << 4);
  if (n0 >= N) return;  // wave-uniform
  float bv = bias ? bias[n0 + (lane & 15)] : 0.f;
  v8f acc;
#pragma unroll
  for (int r = 0; r < 8; ++r) acc[r] = bv;
  for (int kk = 0; kk < K; kk += 32) {
    v16h a = load_frag_a(A, K, m0, kk, lane);
    v16h b = load_frag_b(Bm, K, n0, kk, lane);
    acc = WMMA_F16(a, b, acc);
  }
#pragma unroll
  for (int r = 0; r < 8; ++r) {
    size_t idx = (size_t)(m0 + r + ((lane >> 4) << 3)) * N + n0 + (lane & 15);
    if (Cbf) Cbf[idx] = (f16)acc[r];
    else     Cf[idx]  = acc[r];
  }
}

// ---------------- V^T with zero pad: Vt[b][k][t] ----------------
__global__ void vtrans_kernel(const f16* __restrict__ Vbf, f16* __restrict__ Vt, int n) {
  int i = blockIdx.x * 256 + threadIdx.x;
  if (i < n) {
    int b = i / (DK_ * SP_);
    int r = i % (DK_ * SP_);
    int k = r / SP_, t = r % SP_;
    Vt[i] = (t < S_) ? Vbf[((size_t)b * S_ + t) * DK_ + k] : (f16)0.f;
  }
}

// ---------------- attention raw scores (Q K^T / 8) into final (b,s,h,t) layout ----------
__global__ __launch_bounds__(256) void attn_scores_kernel(
    const f16* __restrict__ Q, const f16* __restrict__ K, float* __restrict__ attn) {
  const int stile = blockIdx.x, h = blockIdx.y, b = blockIdx.z;
  const int lane = threadIdx.x & 31, wave = threadIdx.x >> 5;  // 8 waves
  const int qrow0 = b * S_ + stile * 16;
  v16h a0 = load_frag_a(Q, E_, qrow0, h * DK_ + 0, lane);
  v16h a1 = load_frag_a(Q, E_, qrow0, h * DK_ + 32, lane);
  for (int tt = wave; tt < S_ / 16; tt += 8) {
    const int krow0 = b * S_ + tt * 16;
    v16h b0 = load_frag_b(K, E_, krow0, h * DK_ + 0, lane);
    v16h b1 = load_frag_b(K, E_, krow0, h * DK_ + 32, lane);
    v8f acc;
#pragma unroll
    for (int r = 0; r < 8; ++r) acc[r] = 0.f;
    acc = WMMA_F16(a0, b0, acc);
    acc = WMMA_F16(a1, b1, acc);
#pragma unroll
    for (int r = 0; r < 8; ++r) {
      size_t s = (size_t)b * S_ + stile * 16 + r + ((lane >> 4) << 3);
      attn[(s * NH_ + h) * S_ + tt * 16 + (lane & 15)] = acc[r] * 0.125f;
    }
  }
}

// ---------------- in-place softmax over last dim; export f16 rows for s>=1024 ------------
__global__ __launch_bounds__(128) void attn_softmax_kernel(
    float* __restrict__ attn, f16* __restrict__ Pbf) {
  __shared__ float red[128];
  const int idx = blockIdx.x;               // b*(S*NH) + s*NH + h
  const int h = idx & 3;
  const int s = (idx >> 2) % S_;
  const int b = idx / (S_ * NH_);
  float* row = attn + ((size_t)(b * S_ + s) * NH_ + h) * S_;
  const int tid = threadIdx.x;
  float mx = -1e30f;
  for (int t = tid; t < S_; t += 128) mx = fmaxf(mx, row[t]);
  red[tid] = mx; __syncthreads();
  for (int o = 64; o > 0; o >>= 1) { if (tid < o) red[tid] = fmaxf(red[tid], red[tid + o]); __syncthreads(); }
  mx = red[0]; __syncthreads();
  float sm = 0.f;
  for (int t = tid; t < S_; t += 128) { float e = fast_exp(row[t] - mx); row[t] = e; sm += e; }
  red[tid] = sm; __syncthreads();
  for (int o = 64; o > 0; o >>= 1) { if (tid < o) red[tid] += red[tid + o]; __syncthreads(); }
  float inv = 1.f / red[0];
  f16* pr = (s >= T_) ? (Pbf + (size_t)((b * NH_ + h) * NOUT_ + (s - T_)) * SP_) : nullptr;
  for (int t = tid; t < S_; t += 128) {
    float v = row[t] * inv;
    row[t] = v;
    if (pr) pr[t] = (f16)v;
  }
  if (pr && tid < SP_ - S_) pr[S_ + tid] = (f16)0.f;  // zero K-pad
}

// ---------------- head = P @ V for the last 16 rows only ----------------
__global__ __launch_bounds__(128) void attn_head_kernel(
    const f16* __restrict__ Pbf, const f16* __restrict__ Vt, float* __restrict__ head) {
  const int bh = blockIdx.x;                  // b*4+h
  const int b = bh >> 2;
  const int lane = threadIdx.x & 31, wave = threadIdx.x >> 5;
  const int n0 = wave << 4;                   // 4 waves * 16 = DK
  const f16* Vb = Vt + (size_t)b * DK_ * SP_;
  v8f acc;
#pragma unroll
  for (int r = 0; r < 8; ++r) acc[r] = 0.f;
  for (int kk = 0; kk < SP_; kk += 32) {
    v16h a = load_frag_a(Pbf, SP_, bh * NOUT_, kk, lane);
    v16h bb = load_frag_b(Vb, SP_, n0, kk, lane);
    acc = WMMA_F16(a, bb, acc);
  }
#pragma unroll
  for (int r = 0; r < 8; ++r)
    head[((size_t)bh * NOUT_ + r + ((lane >> 4) << 3)) * DK_ + n0 + (lane & 15)] = acc[r];
}

// ---------------- head mean -> w_h -> final MLP (tiny), 1 block ----------------
__global__ __launch_bounds__(256) void final_kernel(
    const float* __restrict__ head, const float* __restrict__ whW,
    const float* __restrict__ W1, const float* __restrict__ b1,
    const float* __restrict__ W2, const float* __restrict__ b2,
    const float* __restrict__ W3, const float* __restrict__ b3,
    float* __restrict__ scratch, float* __restrict__ fout) {
  const int tid = threadIdx.x;
  float* Ofeat = scratch;                  // 16*4096
  float* h1 = scratch + 16 * 4096;         // 16*64
  float* h2 = h1 + 16 * 64;                // 16*32
  for (int i = tid; i < 16 * 4096; i += 256) {
    int b = i >> 12, c = i & 4095, s = c >> 8, e = c & 255;
    float acc = 0.f;
    for (int k = 0; k < DK_; ++k) {
      float hm = 0.25f * (head[((size_t)((b * 4 + 0) * NOUT_ + s)) * DK_ + k] +
                          head[((size_t)((b * 4 + 1) * NOUT_ + s)) * DK_ + k] +
                          head[((size_t)((b * 4 + 2) * NOUT_ + s)) * DK_ + k] +
                          head[((size_t)((b * 4 + 3) * NOUT_ + s)) * DK_ + k]);
      acc += hm * whW[e * DK_ + k];
    }
    Ofeat[i] = acc;
  }
  __syncthreads();
  for (int i = tid; i < 16 * 64; i += 256) {
    int b = i >> 6, o = i & 63;
    float s = b1[o];
    const float* row = Ofeat + (size_t)b * 4096;
    const float* w = W1 + (size_t)o * 4096;
    for (int j = 0; j < 4096; ++j) s += row[j] * w[j];
    h1[i] = fmaxf(s, 0.f);
  }
  __syncthreads();
  for (int i = tid; i < 16 * 32; i += 256) {
    int b = i >> 5, o = i & 31;
    float s = b2[o];
    for (int j = 0; j < 64; ++j) s += h1[b * 64 + j] * W2[o * 64 + j];
    h2[i] = fmaxf(s, 0.f);
  }
  __syncthreads();
  for (int i = tid; i < 16 * 16; i += 256) {
    int b = i >> 4, o = i & 15;
    float s = b3[o];
    for (int j = 0; j < 32; ++j) s += h2[b * 32 + j] * W3[o * 32 + j];
    fout[i] = s;
  }
}

// ======================= host launcher =======================
extern "C" void kernel_launch(void* const* d_in, const int* in_sizes, int n_in,
                              void* d_out, int out_size, void* d_ws, size_t ws_size,
                              hipStream_t stream) {
  (void)in_sizes; (void)n_in; (void)out_size; (void)ws_size;
  const float* x_tab  = (const float*)d_in[0];
  const float* x_ts   = (const float*)d_in[1];
  const float* x_fut  = (const float*)d_in[2];
  const float* tsW    = (const float*)d_in[3];
  const float* tsb    = (const float*)d_in[4];
  const float* futW   = (const float*)d_in[5];
  const float* futb   = (const float*)d_in[6];
  const float* tabW   = (const float*)d_in[7];
  const float* tabb   = (const float*)d_in[8];
  const float* eWih0  = (const float*)d_in[9];
  const float* eWhh0  = (const float*)d_in[10];
  const float* ebih0  = (const float*)d_in[11];
  const float* ebhh0  = (const float*)d_in[12];
  const float* eWih1  = (const float*)d_in[13];
  const float* eWhh1  = (const float*)d_in[14];
  const float* ebih1  = (const float*)d_in[15];
  const float* ebhh1  = (const float*)d_in[16];
  const float* dWih0  = (const float*)d_in[17];
  const float* dWhh0  = (const float*)d_in[18];
  const float* dbih0  = (const float*)d_in[19];
  const float* dbhh0  = (const float*)d_in[20];
  const float* dWih1  = (const float*)d_in[21];
  const float* dWhh1  = (const float*)d_in[22];
  const float* dbih1  = (const float*)d_in[23];
  const float* dbhh1  = (const float*)d_in[24];
  const float* tabW1  = (const float*)d_in[25];
  const float* tabb1  = (const float*)d_in[26];
  const float* tabW2  = (const float*)d_in[27];
  const float* tabb2  = (const float*)d_in[28];
  const float* tabW3  = (const float*)d_in[29];
  const float* tabb3  = (const float*)d_in[30];
  const float* ln_g   = (const float*)d_in[31];
  const float* ln_b   = (const float*)d_in[32];
  const float* qW     = (const float*)d_in[33];
  const float* kW     = (const float*)d_in[34];
  const float* vW     = (const float*)d_in[35];
  const float* whW    = (const float*)d_in[36];
  const float* lastW1 = (const float*)d_in[37];
  const float* lastb1 = (const float*)d_in[38];
  const float* lastW2 = (const float*)d_in[39];
  const float* lastb2 = (const float*)d_in[40];
  const float* lastW3 = (const float*)d_in[41];
  const float* lastb3 = (const float*)d_in[42];

  char* base = (char*)d_ws;
  size_t off = 0;
  auto alloc = [&](size_t bytes) -> void* {
    off = (off + 255) & ~(size_t)255;
    void* r = base + off;
    off += bytes;
    return r;
  };
  float* st    = (float*)alloc(8192 * 4);
  float* A0    = (float*)alloc(8192 * 4);
  float* c0    = (float*)alloc(1024 * 4);
  float* afut  = (float*)alloc(1024 * 4);
  float* cfut  = (float*)alloc(1024 * 4);
  float* b1e   = (float*)alloc(1024 * 4);
  float* b1d   = (float*)alloc(1024 * 4);
  float* t2    = (float*)alloc(65536 * 4);
  float* mlp1  = (float*)alloc(1024 * 4);
  float* mlp2  = (float*)alloc(512 * 4);
  float* h0e   = (float*)alloc(4096 * 4);
  float* c0e   = (float*)alloc(4096 * 4);
  float* h1e   = (float*)alloc(4096 * 4);
  float* c1e   = (float*)alloc(4096 * 4);
  float* hdmp  = (float*)alloc(4096 * 4);
  float* cdmp  = (float*)alloc(4096 * 4);
  float* headb = (float*)alloc(65536 * 4);
  float* fsc   = (float*)alloc((16 * 4096 + 16 * 64 + 16 * 32) * 4);
  float* xseq  = (float*)alloc((size_t)B_ * S_ * E_ * 4);
  f16* Whh0e   = (f16*)alloc((size_t)G_ * 256 * 2);
  f16* Wp1e    = (f16*)alloc((size_t)G_ * 512 * 2);
  f16* Whh0d   = (f16*)alloc((size_t)G_ * 256 * 2);
  f16* Wp1d    = (f16*)alloc((size_t)G_ * 512 * 2);
  f16* qWb     = (f16*)alloc((size_t)256 * 256 * 2);
  f16* kWb     = (f16*)alloc((size_t)256 * 256 * 2);
  f16* vWb     = (f16*)alloc((size_t)64 * 256 * 2);
  f16* y0b     = (f16*)alloc((size_t)B_ * T_ * E_ * 2);
  f16* z0b     = (f16*)alloc((size_t)B_ * NOUT_ * E_ * 2);
  f16* xnb     = (f16*)alloc((size_t)B_ * S_ * E_ * 2);
  f16* Qb      = (f16*)alloc((size_t)B_ * S_ * E_ * 2);
  f16* Kb      = (f16*)alloc((size_t)B_ * S_ * E_ * 2);
  f16* Vb      = (f16*)alloc((size_t)B_ * S_ * DK_ * 2);
  f16* Vtb     = (f16*)alloc((size_t)B_ * DK_ * SP_ * 2);
  f16* Pb      = (f16*)alloc((size_t)B_ * NH_ * NOUT_ * SP_ * 2);

  float* fout = (float*)d_out;
  float* attn = fout + 256;

  precompute_kernel<<<1, 256, 0, stream>>>(
      x_tab, tabW, tabb, tabW1, tabb1, tabW2, tabb2, tabW3, tabb3,
      tsW, tsb, eWih0, ebih0, ebhh0, futW, futb, dWih0, dbih0, dbhh0,
      ebih1, ebhh1, dbih1, dbhh1,
      t2, mlp1, mlp2, st, A0, c0, afut, cfut, b1e, b1d);

  f32_to_f16_kernel<<<(262144 + 255) / 256, 256, 0, stream>>>(eWhh0, Whh0e, 262144);
  pack_w_kernel<<<(524288 + 255) / 256, 256, 0, stream>>>(eWih1, eWhh1, Wp1e, 256, 524288);
  f32_to_f16_kernel<<<(262144 + 255) / 256, 256, 0, stream>>>(dWhh0, Whh0d, 262144);
  pack_w_kernel<<<(524288 + 255) / 256, 256, 0, stream>>>(dWih1, dWhh1, Wp1d, 256, 524288);
  f32_to_f16_kernel<<<(65536 + 255) / 256, 256, 0, stream>>>(qW, qWb, 65536);
  f32_to_f16_kernel<<<(65536 + 255) / 256, 256, 0, stream>>>(kW, kWb, 65536);
  f32_to_f16_kernel<<<(16384 + 255) / 256, 256, 0, stream>>>(vW, vWb, 16384);

  // encoder L0 (reduced input F=8), init h=c=st[0]
  lstm_seq_kernel<<<1, 512, 0, stream>>>(nullptr, 0, 0, x_ts, 8, T_, A0, c0, Whh0e,
                                         st, st, y0b, T_, nullptr, 0, 0, h0e, c0e);
  // encoder L1 (dense input y0, K=512 packed), init st[1]; writes x_seq rows [0,1024)
  lstm_seq_kernel<<<1, 512, 0, stream>>>(y0b, 256, T_, nullptr, 0, T_, nullptr, b1e, Wp1e,
                                         st + 4096, st + 4096, nullptr, 0, xseq, S_, 0, h1e, c1e);
  // decoder L0 (reduced F=1), init = encoder L0 final state
  lstm_seq_kernel<<<1, 512, 0, stream>>>(nullptr, 0, 0, x_fut, 1, NOUT_, afut, cfut, Whh0d,
                                         h0e, c0e, z0b, NOUT_, nullptr, 0, 0, hdmp, cdmp);
  // decoder L1 (dense z0), init = encoder L1 final state; writes x_seq rows [1024,1040)
  lstm_seq_kernel<<<1, 512, 0, stream>>>(z0b, 256, NOUT_, nullptr, 0, NOUT_, nullptr, b1d, Wp1d,
                                         h1e, c1e, nullptr, 0, xseq, S_, T_, hdmp, cdmp);

  layernorm_kernel<<<B_ * S_, 256, 0, stream>>>(xseq, ln_g, ln_b, xnb);

  gemm_f16_kernel<<<dim3(B_ * S_ / 16, 4), 128, 0, stream>>>(xnb, qWb, nullptr, Qb, nullptr, 256, 256);
  gemm_f16_kernel<<<dim3(B_ * S_ / 16, 4), 128, 0, stream>>>(xnb, kWb, nullptr, Kb, nullptr, 256, 256);
  gemm_f16_kernel<<<dim3(B_ * S_ / 16, 1), 128, 0, stream>>>(xnb, vWb, nullptr, Vb, nullptr, 64, 256);

  vtrans_kernel<<<(B_ * DK_ * SP_ + 255) / 256, 256, 0, stream>>>(Vb, Vtb, B_ * DK_ * SP_);

  attn_scores_kernel<<<dim3(S_ / 16, NH_, B_), 256, 0, stream>>>(Qb, Kb, attn);
  attn_softmax_kernel<<<B_ * S_ * NH_, 128, 0, stream>>>(attn, Pb);
  attn_head_kernel<<<B_ * NH_, 128, 0, stream>>>(Pb, Vtb, headb);

  final_kernel<<<1, 256, 0, stream>>>(headb, whW, lastW1, lastb1, lastW2, lastb2,
                                      lastW3, lastb3, fsc, fout);
}
